// S4Decoder_26998164423382
// MI455X (gfx1250) — matrix operand order repaired and checked by reading, not compiled, not run on hardware
//
#include <hip/hip_runtime.h>
#include <hip/hip_bf16.h>
#include <math.h>

typedef __bf16 bf16;
typedef __attribute__((ext_vector_type(16))) __bf16 v16bf;
typedef __attribute__((ext_vector_type(8)))  float  v8f;

static constexpr int  kB = 32, kH = 256, kL = 1024, kNODE = 512, kN = 128, kL2 = 2048;
#define PI_F 3.14159265358979f

// ---------------- WMMA fragment helpers (contiguous k order: k = 16*(lane/16) + element) ---
__device__ __forceinline__ v16bf load_frag_a(const bf16* t, int ldk) {
  int lane = threadIdx.x & 31;
  int m = lane & 15;
  int koff = (lane >> 4) << 4;                    // lanes 16-31: K += 16
  v16bf a;
#pragma unroll
  for (int v = 0; v < 8; ++v) {
    int kb = 2 * v + koff;
    a[2 * v]     = t[m * ldk + kb];
    a[2 * v + 1] = t[m * ldk + kb + 1];
  }
  return a;
}
__device__ __forceinline__ v16bf load_frag_b(const bf16* t, int ldn) {
  int lane = threadIdx.x & 31;
  int n = lane & 15;
  int koff = (lane >> 4) << 4;                    // lanes 16-31: K += 16 (B layout)
  v16bf b;
#pragma unroll
  for (int v = 0; v < 8; ++v) {
    int kb = 2 * v + koff;
    b[2 * v]     = t[kb * ldn + n];
    b[2 * v + 1] = t[(kb + 1) * ldn + n];
  }
  return b;
}
// Toeplitz B fragment built from a circular k2 row in LDS: T[k][n] = row[(n-k) & 2047]
__device__ __forceinline__ v16bf load_frag_T(const bf16* row, int nbase, int kbase) {
  int lane = threadIdx.x & 31;
  int n = nbase + (lane & 15);
  int koff = (lane >> 4) << 4;
  v16bf b;
#pragma unroll
  for (int v = 0; v < 8; ++v) {
    int kb = kbase + 2 * v + koff;
    b[2 * v]     = row[(n - kb) & (kL2 - 1)];
    b[2 * v + 1] = row[(n - kb - 1) & (kL2 - 1)];
  }
  return b;
}

// ---------------- elementwise converters / small generators ------------------------
__global__ void k_cvt_bf16(const float* __restrict__ in, bf16* __restrict__ out, int n) {
  int i = blockIdx.x * 256 + threadIdx.x;
  if (i < n) out[i] = (bf16)in[i];
}

__global__ void k_gen_cf(bf16* __restrict__ cf) {       // DFT coeff matrix, 256x256 rows: 2f->cos, 2f+1->-sin
  int i = blockIdx.x * 256 + threadIdx.x;
  if (i >= 256 * 256) return;
  int r = i >> 8, t = i & 255;
  float v = 0.f;
  if (r < 254) {
    int f = (r >> 1) + 1;
    float w = 2.f * PI_F * (float)f * (float)t / 256.f;
    v = (r & 1) ? -sinf(w) : cosf(w);
  }
  cf[i] = (bf16)v;
}

// ---------------- time embedding --------------------------------------------------
__global__ void k_silu_emb(const int* __restrict__ t, float* __restrict__ sb) {
  int i = blockIdx.x * 256 + threadIdx.x;              // B*1024
  if (i >= kB * kL) return;
  int b = i >> 10, j = i & 1023;
  int jp = (j < 512) ? j : (j - 512);
  float fr = __expf((float)jp * (-logf(10000.f) / 511.f));
  float te = (float)t[b] * fr;
  float e = (j < 512) ? sinf(te) : cosf(te);
  sb[i] = e / (1.f + __expf(-e));                      // silu
}
__global__ void k_emb_gemm(const float* __restrict__ sb, const float* __restrict__ adaW,
                           const float* __restrict__ adab, float* __restrict__ emb) {
  int i = blockIdx.x * 256 + threadIdx.x;              // B*2048
  if (i >= kB * 2 * kL) return;
  int b = i >> 11, o = i & 2047;
  const float* s = sb + b * kL;
  const float* w = adaW + (size_t)o * kL;
  float acc = adab[o];
  for (int k = 0; k < kL; ++k) acc += s[k] * w[k];
  emb[i] = acc;
}

// ---------------- LN over L with AdaLN scale/shift --------------------------------
__global__ void k_ln_mod(const float* __restrict__ x, const float* __restrict__ emb,
                         float* __restrict__ xmod) {
  __shared__ float red[256];
  int h = blockIdx.x, b = blockIdx.y, tid = threadIdx.x;
  const float* row = x + ((size_t)b * kH + h) * kL;
  float s = 0.f, s2 = 0.f;
  for (int j = tid; j < kL; j += 256) { float v = row[j]; s += v; s2 += v * v; }
  red[tid] = s;  __syncthreads();
  for (int o = 128; o > 0; o >>= 1) { if (tid < o) red[tid] += red[tid + o]; __syncthreads(); }
  float mean = red[0] / kL;  __syncthreads();
  red[tid] = s2; __syncthreads();
  for (int o = 128; o > 0; o >>= 1) { if (tid < o) red[tid] += red[tid + o]; __syncthreads(); }
  float var = red[0] / kL - mean * mean;
  float rq = rsqrtf(var + 1e-5f);
  float* out = xmod + ((size_t)b * kH + h) * kL;
  const float* sc = emb + (size_t)b * 2 * kL;
  for (int j = tid; j < kL; j += 256)
    out[j] = (row[j] - mean) * rq * (1.f + sc[j]) + sc[kL + j];
}

// ---------------- generic bf16 WMMA channel GEMM: out[b,o,l] = sum_ch W[o,ch]X[b,ch,l] ----
// MODE 0: trans  -> outA = acc+bias+add            (h = xmod + trans)
// MODE 1: out    -> u=acc+bias+add; outA=tanh(u)*sig(u); outb=bf16(outA)
// MODE 2: proj   -> o<256: outA=acc+bias ; o>=256: outB=acc+bias, outb=bf16
// MODE 3: dft    -> outA = acc (no bias)
template <int MODE>
__global__ void k_gemm_ch(const bf16* __restrict__ Wb, const bf16* __restrict__ Xb, int O,
                          const float* __restrict__ bias, const float* __restrict__ addf,
                          float* __restrict__ outA, float* __restrict__ outB,
                          bf16* __restrict__ outb) {
  constexpr int CH = 256;
  __shared__ bf16 Wt[32 * 32];
  __shared__ bf16 Xt[32 * 64];
  int lb = blockIdx.x * 64, ob = blockIdx.y * 32, b = blockIdx.z;
  int tid = threadIdx.x, wave = tid >> 5, lane = tid & 31;
  int wm = wave & 1, wl = wave >> 1;
  v8f acc = {};
  for (int kb = 0; kb < CH; kb += 32) {
    for (int i = tid; i < 1024; i += 256) {
      int o = i >> 5, k = i & 31;
      Wt[i] = Wb[(size_t)(ob + o) * CH + kb + k];
    }
    for (int i = tid; i < 2048; i += 256) {
      int k = i >> 6, l = i & 63;
      Xt[i] = Xb[((size_t)b * CH + kb + k) * kL + lb + l];
    }
    __syncthreads();
    v16bf af = load_frag_a(Wt + wm * 16 * 32, 32);
    v16bf bf = load_frag_b(Xt + wl * 16, 64);
    acc = __builtin_amdgcn_wmma_f32_16x16x32_bf16(false, af, false, bf, (short)0, acc, false, false);
    __syncthreads();
  }
  int nn = lane & 15, moff = (lane >> 4) << 3;
  int o0 = ob + wm * 16 + moff, l0 = lb + nn;
#pragma unroll
  for (int r = 0; r < 8; ++r) {
    int o = o0 + r, l = l0;
    float v = acc[r];
    if (MODE != 3) v += bias[o];
    size_t idx = ((size_t)b * O + o) * kL + l;
    if (MODE == 0) {
      outA[idx] = v + addf[idx];
    } else if (MODE == 1) {
      float u = v + addf[idx];
      float w = tanhf(u) * (1.f / (1.f + __expf(-u)));
      outA[idx] = w; outb[idx] = (bf16)w;
    } else if (MODE == 2) {
      if (o < 256) outA[((size_t)b * 256 + o) * kL + l] = v;
      else { size_t j = ((size_t)b * 256 + (o - 256)) * kL + l; outB[j] = v; outb[j] = (bf16)v; }
    } else {
      outA[idx] = v;
    }
  }
}

// ---------------- LN over H (norm_g/norm_b) --------------------------------------
__global__ void k_ln_h(const float* __restrict__ hbuf, const float* __restrict__ g,
                       const float* __restrict__ be, float* __restrict__ zf, bf16* __restrict__ zb) {
  int tid = threadIdx.x, wave = tid >> 5, lane = tid & 31;
  int b = blockIdx.y;
  int l = blockIdx.x * 256 + wave * 32 + lane;
  float s = 0.f, s2 = 0.f;
  const float* base = hbuf + (size_t)b * kH * kL + l;
  for (int h = 0; h < kH; ++h) { float v = base[(size_t)h * kL]; s += v; s2 += v * v; }
  float mean = s / kH, var = s2 / kH - mean * mean;
  float rq = rsqrtf(var + 1e-5f);
  for (int h = 0; h < kH; ++h) {
    float v = (base[(size_t)h * kL] - mean) * rq * g[h] + be[h];
    size_t idx = ((size_t)b * kH + h) * kL + l;
    zf[idx] = v; zb[idx] = (bf16)v;
  }
}

// ---------------- S4 kernel: modal precompute + k2 synthesis ----------------------
__global__ void k_s4_pre(const float* __restrict__ log_dt, const float* __restrict__ Ar,
                         const float* __restrict__ Ai, const float* __restrict__ Cre,
                         const float* __restrict__ Cim, float* __restrict__ cka) {
  int i = blockIdx.x * 256 + threadIdx.x;          // H*N
  if (i >= kH * kN) return;
  int h = i >> 7, n = i & 127;
  float dt = __expf(log_dt[h]);
  float are = -__expf(Ar[i]), aim = Ai[i];
  float dre = dt * are, dim = dt * aim;
  float er = __expf(dre); float sn, cs; __sincosf(dim, &sn, &cs);
  float nre = er * cs - 1.f, nim = er * sn;        // exp(dtA) - 1
  float inv = 1.f / (are * are + aim * aim);
  float qre = (nre * are + nim * aim) * inv;       // (exp(dtA)-1)/A
  float qim = (nim * are - nre * aim) * inv;
  float* o = cka + (size_t)i * 6;
  o[0] = dre; o[1] = dim;
  for (int c = 0; c < 2; ++c) {
    float cr = Cre[(size_t)c * kH * kN + i], ci = Cim[(size_t)c * kH * kN + i];
    o[2 + 2 * c] = cr * qre - ci * qim;
    o[3 + 2 * c] = cr * qim + ci * qre;
  }
}
__global__ void k_s4_k2(const float* __restrict__ cka, bf16* __restrict__ k2) {
  int l = blockIdx.x * 256 + threadIdx.x;
  int h = blockIdx.y;
  if (l >= kL) return;
  float a0 = 0.f, a1 = 0.f;
  const float* base = cka + (size_t)h * kN * 6;
  for (int n = 0; n < kN; ++n) {
    const float* c = base + n * 6;
    float er = __expf(c[0] * (float)l);
    float sn, cs; __sincosf(c[1] * (float)l, &sn, &cs);
    float pr = er * cs, pi = er * sn;
    a0 += c[2] * pr - c[3] * pi;
    a1 += c[4] * pr - c[5] * pi;
  }
  k2[(size_t)h * kL2 + l]              = (bf16)(2.f * a0);  // K0 causal
  k2[(size_t)h * kL2 + (kL2 - 1 - l)]  = (bf16)(2.f * a1);  // K1 reversed into upper half
}

// ---------------- long conv as Toeplitz WMMA GEMM --------------------------------
__global__ void k_conv_wmma(const bf16* __restrict__ zb, const bf16* __restrict__ k2,
                            float* __restrict__ y) {
  __shared__ bf16 row[kL2];
  __shared__ bf16 Zt[32 * 32];
  int h = blockIdx.y, lb = blockIdx.x * 64;
  int tid = threadIdx.x, wave = tid >> 5, lane = tid & 31;
  int wm = wave & 1, wl = wave >> 1;
  for (int i = tid; i < kL2; i += 256) row[i] = k2[(size_t)h * kL2 + i];
  v8f acc = {};
  for (int kb = 0; kb < kL; kb += 32) {
    for (int i = tid; i < 1024; i += 256) {
      int bb = i >> 5, k = i & 31;
      Zt[i] = zb[((size_t)bb * kH + h) * kL + kb + k];
    }
    if (kb + 32 < kL)                                       // global_prefetch_b8 for next z tile
      __builtin_prefetch(zb + ((size_t)(tid >> 3) * kH + h) * kL + kb + 32, 0, 1);
    __syncthreads();
    v16bf af = load_frag_a(Zt + wm * 16 * 32, 32);
    v16bf bf = load_frag_T(row, lb + wl * 16, kb);
    acc = __builtin_amdgcn_wmma_f32_16x16x32_bf16(false, af, false, bf, (short)0, acc, false, false);
    __syncthreads();
  }
  int nn = lane & 15, moff = (lane >> 4) << 3;
  int b0 = wm * 16 + moff, l = lb + wl * 16 + nn;
#pragma unroll
  for (int r = 0; r < 8; ++r)
    y[((size_t)(b0 + r) * kH + h) * kL + l] = acc[r];
}

// ---------------- gelu(y + Dp*z) ---------------------------------------------------
__global__ void k_gelu(float* __restrict__ y, const float* __restrict__ zf,
                       const float* __restrict__ Dp, bf16* __restrict__ gb) {
  int i = blockIdx.x * 256 + threadIdx.x;
  if (i >= kB * kH * kL) return;
  int h = (i >> 10) & 255;
  float u = y[i] + Dp[h] * zf[i];
  float g = 0.5f * u * (1.f + tanhf(0.7978845608f * (u + 0.044715f * u * u * u)));
  y[i] = g; gb[i] = (bf16)g;
}

// ---------------- trend path -------------------------------------------------------
__global__ void k_conv1(const float* __restrict__ o1, const float* __restrict__ W,
                        const float* __restrict__ bi, float* __restrict__ c1) {
  int i = blockIdx.x * 256 + threadIdx.x;          // B*3*L
  if (i >= kB * 3 * kL) return;
  int b = i / (3 * kL), oc = (i / kL) % 3, l = i % kL;
  float acc = bi[oc];
  for (int ic = 0; ic < kH; ++ic) {
    const float* src = o1 + ((size_t)b * kH + ic) * kL;
    const float* w = W + ((size_t)oc * kH + ic) * 3;
    if (l > 0)      acc += src[l - 1] * w[0];
    acc += src[l] * w[1];
    if (l < kL - 1) acc += src[l + 1] * w[2];
  }
  c1[i] = fmaxf(acc, 0.f);
}
__global__ void k_conv2(const float* __restrict__ c1, const float* __restrict__ W,
                        const float* __restrict__ bi, float* __restrict__ c2) {
  int i = blockIdx.x * 256 + threadIdx.x;          // B*NODE*3
  if (i >= kB * kNODE * 3) return;
  int b = i / (kNODE * 3), oc = (i / 3) % kNODE, pos = i % 3;
  float acc = bi[oc];
  for (int ic = 0; ic < kL; ++ic) {
    const float* w = W + ((size_t)oc * kL + ic) * 3;
    for (int kw = 0; kw < 3; ++kw) {
      int pp = pos + kw - 1;
      if (pp >= 0 && pp < 3)
        acc += c1[((size_t)b * 3 + pp) * kL + ic] * w[kw];   // swapaxes(c1,1,2)
    }
  }
  c2[i] = acc;
}
__global__ void k_trend(const float* __restrict__ c2, float* __restrict__ out) {
  int i = blockIdx.x * 256 + threadIdx.x;          // B*H*NODE -> trend flat
  if (i >= kB * kH * kNODE) return;
  int b = i / (kH * kNODE), h = (i / kNODE) % kH, nd = i % kNODE;
  float lin = (float)(h + 1) / (float)(kH + 1);
  const float* c = c2 + ((size_t)b * kNODE + nd) * 3;
  out[i] = c[0] * lin + c[1] * lin * lin + c[2] * lin * lin * lin;
}

// ---------------- season: top-4 of DFT + resynthesis ------------------------------
__global__ void k_topk(const float* __restrict__ Xc, float* __restrict__ tk) {
  int i = blockIdx.x * 256 + threadIdx.x;          // B*1024
  if (i >= kB * kL) return;
  int b = i >> 10, d = i & 1023;
  float mg[4] = {-1.f, -1.f, -1.f, -1.f}, rr[4], ii[4]; int ff[4] = {0, 0, 0, 0};
  for (int f = 0; f < 127; ++f) {
    float re = Xc[((size_t)b * 256 + 2 * f) * kL + d];
    float im = Xc[((size_t)b * 256 + 2 * f + 1) * kL + d];
    float m = re * re + im * im;
    int p = 4;
    while (p > 0 && m > mg[p - 1]) --p;
    if (p < 4) {
      for (int q = 3; q > p; --q) { mg[q] = mg[q-1]; rr[q] = rr[q-1]; ii[q] = ii[q-1]; ff[q] = ff[q-1]; }
      mg[p] = m; rr[p] = re; ii[p] = im; ff[p] = f;
    }
  }
  float* o = tk + (size_t)i * 12;
  for (int k = 0; k < 4; ++k) {
    o[3 * k]     = sqrtf(fmaxf(mg[k], 0.f));
    o[3 * k + 1] = atan2f(ii[k], rr[k]);
    o[3 * k + 2] = (float)(ff[k] + 1) / 256.f;
  }
}
__global__ void k_season(const float* __restrict__ tk, float* __restrict__ out) {
  int i = blockIdx.x * 256 + threadIdx.x;          // B*256*1024 (= B,H,L order)
  if (i >= kB * kH * kL) return;
  int b = i / (kH * kL), t = (i / kL) % kH, d = i % kL;
  const float* o = tk + ((size_t)b * kL + d) * 12;
  float s = 0.f;
  for (int k = 0; k < 4; ++k)
    s += o[3 * k] * cosf(2.f * PI_F * o[3 * k + 2] * (float)t + o[3 * k + 1]);
  out[i] = 2.f * s;                                 // conjugate pair doubles each term
}

// ---------------- mean over H, residual out, linear head --------------------------
__global__ void k_mean_out0(const float* __restrict__ xmod, float* __restrict__ mbuf,
                            float* __restrict__ out0) {
  int b = blockIdx.y;
  int l = blockIdx.x * 256 + threadIdx.x;
  const float* base = xmod + (size_t)b * kH * kL + l;
  float s = 0.f;
  for (int h = 0; h < kH; ++h) s += base[(size_t)h * kL];
  float m = s / kH;
  mbuf[b * kL + l] = m;
  for (int h = 0; h < kH; ++h)
    out0[((size_t)b * kH + h) * kL + l] = base[(size_t)h * kL] - m;
}
__global__ void k_lin(const float* __restrict__ mbuf, const float* __restrict__ W,
                      const float* __restrict__ bi, float* __restrict__ out) {
  int i = blockIdx.x * 256 + threadIdx.x;          // B*NODE
  if (i >= kB * kNODE) return;
  int b = i >> 9, nd = i & 511;
  float acc = bi[nd];
  const float* m = mbuf + b * kL;
  const float* w = W + (size_t)nd * kL;
  for (int l = 0; l < kL; ++l) acc += m[l] * w[l];
  out[i] = acc;
}

// ==================================================================================
extern "C" void kernel_launch(void* const* d_in, const int* in_sizes, int n_in,
                              void* d_out, int out_size, void* d_ws, size_t ws_size,
                              hipStream_t stream) {
  (void)in_sizes; (void)n_in; (void)out_size; (void)ws_size;
  const float* x      = (const float*)d_in[0];
  const int*   t      = (const int*)  d_in[1];
  const float* x_enc  = (const float*)d_in[2];
  const float* ada_W  = (const float*)d_in[3];
  const float* ada_b  = (const float*)d_in[4];
  const float* transW = (const float*)d_in[5];
  const float* transb = (const float*)d_in[6];
  const float* normg  = (const float*)d_in[7];
  const float* normb  = (const float*)d_in[8];
  const float* log_dt = (const float*)d_in[9];
  const float* A_real = (const float*)d_in[10];
  const float* A_imag = (const float*)d_in[11];
  const float* C_re   = (const float*)d_in[12];
  const float* C_im   = (const float*)d_in[13];
  const float* Dp     = (const float*)d_in[14];
  const float* outW   = (const float*)d_in[15];
  const float* outb_p = (const float*)d_in[16];
  const float* projW  = (const float*)d_in[17];
  const float* projb  = (const float*)d_in[18];
  const float* tr1W   = (const float*)d_in[19];
  const float* tr1b   = (const float*)d_in[20];
  const float* tr2W   = (const float*)d_in[21];
  const float* tr2b   = (const float*)d_in[22];
  const float* linW   = (const float*)d_in[23];
  const float* linb   = (const float*)d_in[24];

  char* ws = (char*)d_ws;
  const size_t BHL = (size_t)kB * kH * kL;                 // 8388608
  float* xmod  = (float*)(ws + 0);                         // 32MB (live to end)
  float* hbuf  = (float*)(ws + 33554432);                  // 32MB ; later reused as Xc (DFT)
  float* zf    = (float*)(ws + 67108864);                  // 32MB ; later reused as outF
  bf16*  zb    = (bf16*) (ws + 100663296);                 // 16MB
  bf16*  xeb   = (bf16*) (ws + 117440512);                 // 16MB ; later reused as outB16
  float* yf    = (float*)(ws + 134217728);                 // 32MB (y/g) ; later reused as out1
  bf16*  gb    = (bf16*) (ws + 167772160);                 // 16MB ; later reused as out2 bf16
  float* out2f = (float*)(ws + 184549376);                 // 32MB
  bf16*  k2b   = (bf16*) (ws + 218103808);                 // 1MB
  bf16*  wtb   = (bf16*) (ws + 219152384);
  bf16*  wob   = (bf16*) (ws + 219283456);
  bf16*  wpb   = (bf16*) (ws + 219414528);
  bf16*  cfb   = (bf16*) (ws + 219676672);
  float* sb    = (float*)(ws + 219807744);
  float* emb   = (float*)(ws + 219938816);
  float* cka   = (float*)(ws + 220200960);
  float* c1    = (float*)(ws + 220987392);
  float* c2    = (float*)(ws + 221380608);
  float* tk    = (float*)(ws + 221577216);
  float* mbuf  = (float*)(ws + 223150080);

  float* Xc     = hbuf;                                    // alias after h dead
  float* outF   = zf;                                      // alias after z dead
  bf16*  outB16 = xeb;                                     // alias after x_enc bf16 dead
  float* out1   = yf;                                      // alias after g dead
  bf16*  out2b  = gb;                                      // alias after gb dead

  float* o_res    = (float*)d_out;                         // (B,H,L)
  float* o_lin    = o_res + BHL;                           // (B,NODE)
  float* o_trend  = o_lin + (size_t)kB * kNODE;            // (B,H,NODE)
  float* o_season = o_trend + (size_t)kB * kH * kNODE;     // (B,H,L)

  dim3 blk(256);
  // precision conversions + coefficient generation
  k_cvt_bf16<<<(BHL + 255) / 256, blk, 0, stream>>>(x_enc, xeb, (int)BHL);
  k_cvt_bf16<<<(kH * kH + 255) / 256, blk, 0, stream>>>(transW, wtb, kH * kH);
  k_cvt_bf16<<<(kH * kH + 255) / 256, blk, 0, stream>>>(outW, wob, kH * kH);
  k_cvt_bf16<<<(2 * kH * kH + 255) / 256, blk, 0, stream>>>(projW, wpb, 2 * kH * kH);
  k_gen_cf<<<(256 * 256 + 255) / 256, blk, 0, stream>>>(cfb);

  // adaLN time embedding
  k_silu_emb<<<(kB * kL + 255) / 256, blk, 0, stream>>>(t, sb);
  k_emb_gemm<<<(kB * 2 * kL + 255) / 256, blk, 0, stream>>>(sb, ada_W, ada_b, emb);
  k_ln_mod<<<dim3(kH, kB), blk, 0, stream>>>(x, emb, xmod);

  // h = xmod + trans_W @ x_enc + trans_b   (WMMA)
  k_gemm_ch<0><<<dim3(kL / 64, kH / 32, kB), blk, 0, stream>>>(
      wtb, xeb, kH, transb, xmod, hbuf, nullptr, nullptr);

  // z = LN_H(h)
  k_ln_h<<<dim3(kL / 256, kB), blk, 0, stream>>>(hbuf, normg, normb, zf, zb);

  // S4 kernel -> k2 (bf16 circular row per h)
  k_s4_pre<<<(kH * kN + 255) / 256, blk, 0, stream>>>(log_dt, A_real, A_imag, C_re, C_im, cka);
  k_s4_k2<<<dim3(kL / 256, kH), blk, 0, stream>>>(cka, k2b);

  // y = circular conv(z, k2)[:L] as Toeplitz WMMA GEMM
  k_conv_wmma<<<dim3(kL / 64, kH), blk, 0, stream>>>(zb, k2b, yf);

  // g = gelu(y + Dp*z)
  k_gelu<<<(BHL + 255) / 256, blk, 0, stream>>>(yf, zf, Dp, gb);

  // out = tanh(u)*sigmoid(u), u = out_W@g + out_b + h   (WMMA)
  k_gemm_ch<1><<<dim3(kL / 64, kH / 32, kB), blk, 0, stream>>>(
      wob, gb, kH, outb_p, hbuf, outF, nullptr, outB16);

  // p = proj_W@out + proj_b -> out1 (f32), out2 (f32+bf16)   (WMMA)
  k_gemm_ch<2><<<dim3(kL / 64, 2 * kH / 32, kB), blk, 0, stream>>>(
      wpb, outB16, 2 * kH, projb, nullptr, out1, out2f, out2b);

  // trend branch
  k_conv1<<<(kB * 3 * kL + 255) / 256, blk, 0, stream>>>(out1, tr1W, tr1b, c1);
  k_conv2<<<(kB * kNODE * 3 + 255) / 256, blk, 0, stream>>>(c1, tr2W, tr2b, c2);
  k_trend<<<(kB * kH * kNODE + 255) / 256, blk, 0, stream>>>(c2, o_trend);

  // season branch: DFT over H as WMMA GEMM, then top-4 + resynthesis
  k_gemm_ch<3><<<dim3(kL / 64, kH / 32, kB), blk, 0, stream>>>(
      cfb, out2b, kH, nullptr, nullptr, Xc, nullptr, nullptr);
  k_topk<<<(kB * kL + 255) / 256, blk, 0, stream>>>(Xc, tk);
  k_season<<<(BHL + 255) / 256, blk, 0, stream>>>(tk, o_season);

  // residual mean + linear head
  k_mean_out0<<<dim3(kL / 256, kB), blk, 0, stream>>>(xmod, mbuf, o_res);
  k_lin<<<(kB * kNODE + 255) / 256, blk, 0, stream>>>(mbuf, linW, linb, o_lin);
}
